// BERT_22282290331966
// MI455X (gfx1250) — matrix operand order repaired and checked by reading, not compiled, not run on hardware
//
#include <hip/hip_runtime.h>

// ---------------- CDNA5 / gfx1250 WMMA types ----------------
typedef __attribute__((ext_vector_type(16))) __bf16 v16bf;
typedef __attribute__((ext_vector_type(8)))  float  v8f;
typedef __attribute__((ext_vector_type(4)))  unsigned int u32x4;

static constexpr int Lc = 12, Hc = 12, Ec = 768, Dc = 64, Fc = 3072, Tc = 513, Bc = 16;
static constexpr int Mrows = Bc * Tc; // 8208

__device__ inline unsigned short f2bf(float f) {
  unsigned int u = __float_as_uint(f);
  u += 0x7fffu + ((u >> 16) & 1u);   // round-to-nearest-even
  return (unsigned short)(u >> 16);
}

union Frag {
  v16bf v;
  u32x4 q[2];
  unsigned short h[16];
};
union Pack8 { u32x4 q; unsigned short h[8]; };

__device__ inline v8f vzero() {
  v8f z = {0.f, 0.f, 0.f, 0.f, 0.f, 0.f, 0.f, 0.f};
  return z;
}

__device__ inline v8f wmma_bf16(v16bf a, v16bf b, v8f c) {
  // D = A(16x32 bf16) * B(32x16 bf16) + C(16x16 f32)
  return __builtin_amdgcn_wmma_f32_16x16x32_bf16(false, a, false, b, (short)0, c, false, false);
}

// ---------------- embeddings: h = tok_emb[x] + pos_emb[t] + seg_emb[0] ----------------
__global__ void embed_kernel(const int* __restrict__ x, const float* __restrict__ tok,
                             const float* __restrict__ pos, const float* __restrict__ seg,
                             float* __restrict__ h) {
  int idx = blockIdx.x * 256 + threadIdx.x;
  if (idx >= Mrows * Ec) return;
  int row = idx / Ec, e = idx - row * Ec;
  int t = row % Tc;
  h[idx] = tok[(size_t)x[row] * Ec + e] + pos[(size_t)t * Ec + e] + seg[e];
}

// ---------------- LayerNorm (fp32 in -> bf16 out) ----------------
__global__ __launch_bounds__(256) void ln_kernel(const float* __restrict__ x,
                                                 const float* __restrict__ gm,
                                                 const float* __restrict__ bt,
                                                 unsigned short* __restrict__ out) {
  int row = blockIdx.x, tid = threadIdx.x;
  const float* xr = x + (size_t)row * Ec;
  float v0 = xr[tid], v1 = xr[tid + 256], v2 = xr[tid + 512];
  float s = v0 + v1 + v2, s2 = v0 * v0 + v1 * v1 + v2 * v2;
  __shared__ float rA[256], rB[256];
  rA[tid] = s; rB[tid] = s2;
  __syncthreads();
  for (int off = 128; off > 0; off >>= 1) {
    if (tid < off) { rA[tid] += rA[tid + off]; rB[tid] += rB[tid + off]; }
    __syncthreads();
  }
  float mean = rA[0] * (1.0f / Ec);
  float var  = rB[0] * (1.0f / Ec) - mean * mean;
  float rstd = rsqrtf(var + 1e-5f);
  unsigned short* orow = out + (size_t)row * Ec;
  orow[tid]       = f2bf((v0 - mean) * rstd * gm[tid]       + bt[tid]);
  orow[tid + 256] = f2bf((v1 - mean) * rstd * gm[tid + 256] + bt[tid + 256]);
  orow[tid + 512] = f2bf((v2 - mean) * rstd * gm[tid + 512] + bt[tid + 512]);
}

// ---------------- weight packing (fp32 -> bf16) ----------------
// W[h,e,d] -> B[e, h*64+d]
__global__ void pack_qkv_kernel(const float* __restrict__ w, unsigned short* __restrict__ o) {
  int idx = blockIdx.x * 256 + threadIdx.x;
  if (idx >= Ec * Ec) return;
  int e = idx / Ec, c = idx - e * Ec;
  o[idx] = f2bf(w[((size_t)(c >> 6) * Ec + e) * Dc + (c & 63)]);
}
__global__ void pack_cvt_kernel(const float* __restrict__ w, unsigned short* __restrict__ o, int n) {
  int idx = blockIdx.x * 256 + threadIdx.x;
  if (idx >= n) return;
  o[idx] = f2bf(w[idx]);
}

// ---------------- tiled WMMA GEMM: C = A(bf16,[M,K]) * B(bf16,[K,N]) + bias ----------------
// Block tile: 128(M) x 64(N), 4 waves, each wave 32x64 (8 wmma / k-step).
// Software-pipelined: register-staged double-buffered LDS for B, prefetched A fragments.
// MODE 0: bf16 out   MODE 1: bf16 out + ReLU   MODE 2: fp32 out, in-place residual add
template <int MODE>
__global__ __launch_bounds__(128) void gemm_kernel(const unsigned short* __restrict__ A,
                                                   const unsigned short* __restrict__ Bm,
                                                   const float* __restrict__ bias,
                                                   void* __restrict__ Cout,
                                                   int M, int N, int K) {
  __shared__ __align__(16) unsigned short BT[2][64][72]; // double-buffered transposed B tile [n][k]
  const int n0 = blockIdx.x * 64, m0 = blockIdx.y * 128;
  const int tid = threadIdx.x, wave = tid >> 5, lane = tid & 31;
  const int g = lane >> 4, ln_ = lane & 15;
  const int kk = tid >> 2, nq = tid & 3; // cooperative B-tile roles

  v8f acc[2][4] = {{vzero(), vzero(), vzero(), vzero()},
                   {vzero(), vzero(), vzero(), vzero()}};

  int ar0 = m0 + wave * 32 + ln_;      if (ar0 > M - 1) ar0 = M - 1;
  int ar1 = m0 + wave * 32 + 16 + ln_; if (ar1 > M - 1) ar1 = M - 1;
  const unsigned short* ap0 = A + (size_t)ar0 * K;
  const unsigned short* ap1 = A + (size_t)ar1 * K;
  const unsigned short* brow = Bm + (size_t)kk * N + n0 + nq * 16;
  const int nsteps = K >> 5;

  // ---- prologue: stage tile 0 into registers, load A fragments for k-step 0 ----
  Pack8 br0, br1;
  br0.q = *(const u32x4*)brow;
  br1.q = *(const u32x4*)(brow + 8);
  Frag af0, af1;
  af0.q[0] = *(const u32x4*)(ap0 + 8 * g);
  af0.q[1] = *(const u32x4*)(ap0 + 16 + 8 * g);
  af1.q[0] = *(const u32x4*)(ap1 + 8 * g);
  af1.q[1] = *(const u32x4*)(ap1 + 16 + 8 * g);

  for (int ks = 0; ks < nsteps; ++ks) {
    const int buf = ks & 1;
    // commit staged tile registers into this iteration's LDS buffer
#pragma unroll
    for (int i = 0; i < 8; ++i) {
      BT[buf][nq * 16 + i][kk]     = br0.h[i];
      BT[buf][nq * 16 + 8 + i][kk] = br1.h[i];
    }
    __syncthreads();

    // issue next tile's global loads (in flight during this iteration's compute)
    const int kn = (ks + 1 < nsteps) ? (ks + 1) : ks; // branchless tail (redundant last reload)
    {
      const unsigned short* bs = brow + (size_t)(kn * 32) * N;
      br0.q = *(const u32x4*)bs;
      br1.q = *(const u32x4*)(bs + 8);
      __builtin_prefetch(bs + (size_t)32 * N, 0, 1); // global_prefetch_b8, one tile further
    }
    // prefetch next A fragments
    Frag an0, an1;
    {
      const unsigned short* ap = ap0 + kn * 32;
      an0.q[0] = *(const u32x4*)(ap + 8 * g);
      an0.q[1] = *(const u32x4*)(ap + 16 + 8 * g);
      ap = ap1 + kn * 32;
      an1.q[0] = *(const u32x4*)(ap + 8 * g);
      an1.q[1] = *(const u32x4*)(ap + 16 + 8 * g);
    }

#pragma unroll
    for (int j = 0; j < 4; ++j) {
      // B-fragment: lane holds column n, k = {8g..8g+7, 16+8g..16+8g+7} from BT row
      Frag bfr;
      const unsigned short* bp = &BT[buf][j * 16 + ln_][8 * g];
      bfr.q[0] = *(const u32x4*)bp;
      bfr.q[1] = *(const u32x4*)(bp + 16);
      acc[0][j] = wmma_bf16(af0.v, bfr.v, acc[0][j]);
      acc[1][j] = wmma_bf16(af1.v, bfr.v, acc[1][j]);
    }
    af0 = an0;
    af1 = an1;
  }

#pragma unroll
  for (int s = 0; s < 2; ++s) {
#pragma unroll
    for (int j = 0; j < 4; ++j) {
      const int col = n0 + j * 16 + ln_;
      const float bv = bias[col];
#pragma unroll
      for (int r = 0; r < 8; ++r) {
        const int row = m0 + wave * 32 + s * 16 + r + 8 * g; // C: VGPR r -> M=r+8g, N=lane&15
        if (row < M) {
          float val = acc[s][j][r] + bv;
          if (MODE == 1) val = fmaxf(val, 0.f);
          if (MODE == 2) ((float*)Cout)[(size_t)row * N + col] += val;
          else           ((unsigned short*)Cout)[(size_t)row * N + col] = f2bf(val);
        }
      }
    }
  }
}

// ---------------- attention per (b,h): S^T = Q*K^T computed transposed ----------------
// Two-pass flash: pass A accumulates per-t (per-lane) max m and sum l (scalar only);
// pass B recomputes S^T, P = exp(S - m - log l) lands directly in the A-fragment
// layout for O = P*V (no LDS transpose, no rescale, no end division).
__global__ __launch_bounds__(128) void attn_kernel(const unsigned short* __restrict__ Q,
                                                   const unsigned short* __restrict__ Kb,
                                                   const unsigned short* __restrict__ V,
                                                   unsigned short* __restrict__ O) {
  __shared__ __align__(16) unsigned short VT[64][72]; // V tile transposed [d][s], shared by all waves
  const int tid = threadIdx.x, wave = tid >> 5, lane = tid & 31;
  const int g = lane >> 4, ln_ = lane & 15;
  const int bh = blockIdx.y, b = bh / Hc, hh = bh - b * Hc;
  const int t0 = blockIdx.x * 64 + wave * 16; // this wave's 16 t-columns
  const size_t rowbase = (size_t)b * Tc;
  const int colbase = hh * Dc;
  const int nsteps = (Tc + 31) / 32; // 17, uniform

  // K supplies B columns t: contiguous d in memory -> direct 16B loads, loaded once
  Frag kb[2];
  {
    int tr = t0 + ln_; if (tr > Tc - 1) tr = Tc - 1;
    const unsigned short* kp = Kb + (rowbase + tr) * Ec + colbase;
#pragma unroll
    for (int ds = 0; ds < 2; ++ds) {
      kb[ds].q[0] = *(const u32x4*)(kp + 32 * ds + 8 * g);
      kb[ds].q[1] = *(const u32x4*)(kp + 32 * ds + 16 + 8 * g);
    }
  }

  float m = -3.0e38f, l = 0.f;

  // ---- pass A: max + sum over s (per lane; s lives in the VGPR dimension) ----
  for (int ss = 0; ss < nsteps; ++ss) {
    const int s0 = ss * 32;
    v8f S[2] = {vzero(), vzero()};
#pragma unroll
    for (int sh = 0; sh < 2; ++sh) {
      int srow = s0 + 16 * sh + ln_; if (srow > Tc - 1) srow = Tc - 1;
      const unsigned short* qp = Q + (rowbase + srow) * Ec + colbase;
#pragma unroll
      for (int ds = 0; ds < 2; ++ds) {
        Frag qa;
        qa.q[0] = *(const u32x4*)(qp + 32 * ds + 8 * g);
        qa.q[1] = *(const u32x4*)(qp + 32 * ds + 16 + 8 * g);
        S[sh] = wmma_bf16(qa.v, kb[ds].v, S[sh]);
      }
    }
    float sv[16];
    float lm = -3.0e38f;
#pragma unroll
    for (int r = 0; r < 8; ++r) {
      const int sA = s0 + r + 8 * g;       // S[0] row -> s index
      float a0 = (sA < Tc)      ? S[0][r] : -3.0e38f;
      float a1 = (sA + 16 < Tc) ? S[1][r] : -3.0e38f;
      sv[r] = a0; sv[8 + r] = a1;
      lm = fmaxf(lm, fmaxf(a0, a1));
    }
    lm = fmaxf(lm, __shfl_xor(lm, 16));    // combine the two s-halves held by partner lane
    const float nm = fmaxf(m, lm);
    const float corr = __expf(m - nm);
    float ps = 0.f;
#pragma unroll
    for (int i = 0; i < 16; ++i) ps += __expf(sv[i] - nm);
    ps += __shfl_xor(ps, 16);
    l = l * corr + ps;
    m = nm;
  }
  const float c = m + __logf(l); // P = exp(S - c) is fully normalized

  // ---- pass B: recompute S^T, P -> A-fragment, O += P*V ----
  v8f oacc[4] = {vzero(), vzero(), vzero(), vzero()};
  for (int ss = 0; ss < nsteps; ++ss) {
    const int s0 = ss * 32;
    v8f S[2] = {vzero(), vzero()};
#pragma unroll
    for (int sh = 0; sh < 2; ++sh) {
      int srow = s0 + 16 * sh + ln_; if (srow > Tc - 1) srow = Tc - 1;
      const unsigned short* qp = Q + (rowbase + srow) * Ec + colbase;
#pragma unroll
      for (int ds = 0; ds < 2; ++ds) {
        Frag qa;
        qa.q[0] = *(const u32x4*)(qp + 32 * ds + 8 * g);
        qa.q[1] = *(const u32x4*)(qp + 32 * ds + 16 + 8 * g);
        S[sh] = wmma_bf16(qa.v, kb[ds].v, S[sh]);
      }
    }
    // P values land exactly in A-fragment K-order: g=0 -> s {0..7,16..23}, g=1 -> {8..15,24..31}
    Frag pf;
#pragma unroll
    for (int r = 0; r < 8; ++r) {
      const int sA = s0 + r + 8 * g;
      float a0 = (sA < Tc)      ? S[0][r] : -3.0e38f;
      float a1 = (sA + 16 < Tc) ? S[1][r] : -3.0e38f;
      pf.h[r]     = f2bf(__expf(a0 - c));
      pf.h[8 + r] = f2bf(__expf(a1 - c));
    }

    __syncthreads();
    { // cooperative V tile (32 s x 64 d) transposed into LDS, shared by all 4 waves
      int kk = tid >> 2, nq = tid & 3;
      int srow = s0 + kk; if (srow > Tc - 1) srow = Tc - 1;
      const unsigned short* vp = V + (rowbase + srow) * Ec + colbase + nq * 16;
      Pack8 p0, p1;
      p0.q = *(const u32x4*)vp;
      p1.q = *(const u32x4*)(vp + 8);
#pragma unroll
      for (int i = 0; i < 8; ++i) {
        VT[nq * 16 + i][kk]     = p0.h[i];
        VT[nq * 16 + 8 + i][kk] = p1.h[i];
      }
    }
    __syncthreads();

#pragma unroll
    for (int j = 0; j < 4; ++j) {
      Frag vb; // B[k=s][n=d]: contiguous s from VT row d
      const unsigned short* bp = &VT[16 * j + ln_][8 * g];
      vb.q[0] = *(const u32x4*)bp;
      vb.q[1] = *(const u32x4*)(bp + 16);
      oacc[j] = wmma_bf16(pf.v, vb.v, oacc[j]);
    }
  }

#pragma unroll
  for (int j = 0; j < 4; ++j) {
#pragma unroll
    for (int r = 0; r < 8; ++r) {
      const int t = t0 + r + 8 * g; // O rows are t (M = r + 8g)
      if (t < Tc) {
        O[(rowbase + t) * Ec + colbase + 16 * j + ln_] = f2bf(oacc[j][r]);
      }
    }
  }
}

// ---------------- host orchestration ----------------
extern "C" void kernel_launch(void* const* d_in, const int* in_sizes, int n_in,
                              void* d_out, int out_size, void* d_ws, size_t ws_size,
                              hipStream_t stream) {
  (void)in_sizes; (void)n_in; (void)out_size;
  const int*   x   = (const int*)d_in[0];
  const float* tok = (const float*)d_in[1];
  const float* pos = (const float*)d_in[2];
  const float* seg = (const float*)d_in[3];
  const float* Wk  = (const float*)d_in[4];
  const float* Wq  = (const float*)d_in[5];
  const float* Wv  = (const float*)d_in[6];
  const float* bk  = (const float*)d_in[7];
  const float* bq  = (const float*)d_in[8];
  const float* bv  = (const float*)d_in[9];
  const float* Wo  = (const float*)d_in[10];
  const float* bo  = (const float*)d_in[11];
  const float* g1  = (const float*)d_in[12];
  const float* be1 = (const float*)d_in[13];
  const float* g2  = (const float*)d_in[14];
  const float* be2 = (const float*)d_in[15];
  const float* W1  = (const float*)d_in[16];
  const float* b1  = (const float*)d_in[17];
  const float* W2  = (const float*)d_in[18];
  const float* b2  = (const float*)d_in[19];
  float* h = (float*)d_out;

  char* wsb = (char*)d_ws;
  size_t off = 0;
  auto carve = [&](size_t bytes) -> void* {
    void* p = wsb + off;
    off += (bytes + 255) & ~(size_t)255;
    return p;
  };
  unsigned short* nbuf = (unsigned short*)carve((size_t)Mrows * Ec * 2);
  unsigned short* qbuf = (unsigned short*)carve((size_t)Mrows * Ec * 2);
  unsigned short* kbuf = (unsigned short*)carve((size_t)Mrows * Ec * 2);
  unsigned short* vbuf = (unsigned short*)carve((size_t)Mrows * Ec * 2);
  unsigned short* obuf = (unsigned short*)carve((size_t)Mrows * Ec * 2);
  unsigned short* mbuf = (unsigned short*)carve((size_t)Mrows * Ec * 2);
  unsigned short* fbuf = (unsigned short*)carve((size_t)Mrows * Fc * 2);
  unsigned short* wqp  = (unsigned short*)carve((size_t)Ec * Ec * 2);
  unsigned short* wkp  = (unsigned short*)carve((size_t)Ec * Ec * 2);
  unsigned short* wvp  = (unsigned short*)carve((size_t)Ec * Ec * 2);
  unsigned short* wop  = (unsigned short*)carve((size_t)Ec * Ec * 2);
  unsigned short* w1p  = (unsigned short*)carve((size_t)Ec * Fc * 2);
  unsigned short* w2p  = (unsigned short*)carve((size_t)Fc * Ec * 2);
  if (off > ws_size) return;

  const dim3 blk128(128), blk256(256);
  const dim3 gE((Ec * Ec + 255) / 256);
  const dim3 gF((Ec * Fc + 255) / 256);
  const dim3 gGemmE(Ec / 64, (Mrows + 127) / 128);
  const dim3 gGemmF(Fc / 64, (Mrows + 127) / 128);
  const dim3 gAttn((Tc + 63) / 64, Bc * Hc);

  embed_kernel<<<(Mrows * Ec + 255) / 256, blk256, 0, stream>>>(x, tok, pos, seg, h);

  for (int l = 0; l < Lc; ++l) {
    const size_t wqkv = (size_t)l * Hc * Ec * Dc;
    pack_qkv_kernel<<<gE, blk256, 0, stream>>>(Wq + wqkv, wqp);
    pack_qkv_kernel<<<gE, blk256, 0, stream>>>(Wk + wqkv, wkp);
    pack_qkv_kernel<<<gE, blk256, 0, stream>>>(Wv + wqkv, wvp);
    pack_cvt_kernel<<<gE, blk256, 0, stream>>>(Wo + (size_t)l * Ec * Ec, wop, Ec * Ec);
    pack_cvt_kernel<<<gF, blk256, 0, stream>>>(W1 + (size_t)l * Ec * Fc, w1p, Ec * Fc);
    pack_cvt_kernel<<<gF, blk256, 0, stream>>>(W2 + (size_t)l * Fc * Ec, w2p, Fc * Ec);

    ln_kernel<<<Mrows, blk256, 0, stream>>>(h, g1 + (size_t)l * Ec, be1 + (size_t)l * Ec, nbuf);

    gemm_kernel<0><<<gGemmE, blk128, 0, stream>>>(nbuf, wqp, bq + (size_t)l * Ec, qbuf, Mrows, Ec, Ec);
    gemm_kernel<0><<<gGemmE, blk128, 0, stream>>>(nbuf, wkp, bk + (size_t)l * Ec, kbuf, Mrows, Ec, Ec);
    gemm_kernel<0><<<gGemmE, blk128, 0, stream>>>(nbuf, wvp, bv + (size_t)l * Ec, vbuf, Mrows, Ec, Ec);

    attn_kernel<<<gAttn, blk128, 0, stream>>>(qbuf, kbuf, vbuf, obuf);

    gemm_kernel<2><<<gGemmE, blk128, 0, stream>>>(obuf, wop, bo + (size_t)l * Ec, h, Mrows, Ec, Ec);

    ln_kernel<<<Mrows, blk256, 0, stream>>>(h, g2 + (size_t)l * Ec, be2 + (size_t)l * Ec, mbuf);

    gemm_kernel<1><<<gGemmF, blk128, 0, stream>>>(mbuf, w1p, b1 + (size_t)l * Fc, fbuf, Mrows, Fc, Ec);
    gemm_kernel<2><<<gGemmE, blk128, 0, stream>>>(fbuf, w2p, b2 + (size_t)l * Ec, h, Mrows, Ec, Fc);
  }
}
